// EncdecMultiheadAttn_23407571763296
// MI455X (gfx1250) — matrix operand order repaired and checked by reading, not compiled, not run on hardware
//
#include <hip/hip_runtime.h>
#include <math.h>

// ---------------------------------------------------------------------------
// Types for CDNA5 WMMA (wave32)
// ---------------------------------------------------------------------------
typedef __bf16 bf16;
typedef __attribute__((ext_vector_type(8)))  __bf16 v8bf;
typedef __attribute__((ext_vector_type(16))) __bf16 v16bf;
typedef __attribute__((ext_vector_type(8)))  float  v8f;

union V16 { v16bf v; v8bf h[2]; };

#define LQ 1024
#define LK 1024
#define BB 8
#define EE 1024
#define HH 16
#define DD 64

// ---------------------------------------------------------------------------
// f32 -> bf16 conversion (grid-stride)
// ---------------------------------------------------------------------------
__global__ void cvt_f32_bf16(const float* __restrict__ in, bf16* __restrict__ out, int n) {
    for (int i = blockIdx.x * blockDim.x + threadIdx.x; i < n; i += gridDim.x * blockDim.x)
        out[i] = (bf16)in[i];
}

// ---------------------------------------------------------------------------
// Generic TN GEMM:  C[m,n] = scale * sum_k A[m,k] * W[n,k]
// A: M x K bf16 row-major (lda = K), W: N x K bf16 row-major (ldb = K)
// C: M x N, bf16 or f32 (out_f32 flag), ldc = N
// Block: 256 threads = 8 waves arranged 4(M) x 2(N).
// Wave tile 32x64 -> 2x4 WMMA tiles (8 v_wmma per 32-wide K step).
// Block tile 128 x 128.
//
// Per-lane fragment addressing follows ISA 7.12.2:
//  A (16-bit 16x32): lane<16 -> row=lane, K {0..7,16..23}; lane>=16 -> K {8..15,24..31}
//  B (16-bit 32x16): lane<16 -> col=lane, K {0..7,16..23}; lane>=16 -> K {8..15,24..31}
//  C (f32 16x16):    lane<16 -> N=lane, M=r; lane>=16 -> N=lane-16, M=8+r
// ---------------------------------------------------------------------------
__global__ __launch_bounds__(256)
void gemm_bf16_tn(const bf16* __restrict__ A, const bf16* __restrict__ W,
                  void* __restrict__ C, int M, int N, int K,
                  float scale, int out_f32)
{
    const int lane = threadIdx.x & 31;
    const int wave = threadIdx.x >> 5;
    const int wm = wave & 3;           // 0..3
    const int wn = wave >> 2;          // 0..1
    const int m0 = blockIdx.x * 128 + wm * 32;
    const int n0 = blockIdx.y * 128 + wn * 64;
    const int lr = lane & 15;
    const int hi = lane >> 4;

    v8f acc[2][4] = {};

    for (int k0 = 0; k0 < K; k0 += 32) {
        V16 a[2], b[4];
#pragma unroll
        for (int mi = 0; mi < 2; ++mi) {
            const bf16* p = A + (size_t)(m0 + mi * 16 + lr) * K + k0 + hi * 8;
            a[mi].h[0] = *(const v8bf*)(p);
            a[mi].h[1] = *(const v8bf*)(p + 16);
        }
#pragma unroll
        for (int ni = 0; ni < 4; ++ni) {
            const bf16* p = W + (size_t)(n0 + ni * 16 + lr) * K + k0 + hi * 8;
            b[ni].h[0] = *(const v8bf*)(p);
            b[ni].h[1] = *(const v8bf*)(p + 16);
        }
#pragma unroll
        for (int mi = 0; mi < 2; ++mi)
#pragma unroll
            for (int ni = 0; ni < 4; ++ni)
                acc[mi][ni] = __builtin_amdgcn_wmma_f32_16x16x32_bf16(
                    false, a[mi].v, false, b[ni].v, (short)0, acc[mi][ni], false, false);
    }

#pragma unroll
    for (int mi = 0; mi < 2; ++mi) {
#pragma unroll
        for (int ni = 0; ni < 4; ++ni) {
            const int col = n0 + ni * 16 + lr;
#pragma unroll
            for (int r = 0; r < 8; ++r) {
                const int row = m0 + mi * 16 + hi * 8 + r;
                const float v = acc[mi][ni][r] * scale;
                if (out_f32)
                    ((float*)C)[(size_t)row * N + col] = v;
                else
                    ((bf16*)C)[(size_t)row * N + col] = (bf16)v;
            }
        }
    }
}

// ---------------------------------------------------------------------------
// Transpose V out of the KV projection into [b,h][d][LK] bf16 so the P@V
// WMMA B fragments become two contiguous global_load_b128 per lane.
// Block: one (b,h) x 64-row L tile; LDS-tiled 64x64 transpose.
// ---------------------------------------------------------------------------
#define TSTR 72
__global__ __launch_bounds__(256)
void transpose_v(const bf16* __restrict__ kvp, bf16* __restrict__ vT)
{
    __shared__ bf16 tile[64 * TSTR];

    const int bh = blockIdx.x;
    const int h = bh & (HH - 1);
    const int b = bh >> 4;
    const int l0 = blockIdx.y * 64;
    const size_t kstride = (size_t)BB * 2 * EE;    // 16384

    // read: 64 L rows x 64 d, coalesced along d
    {
        const int ll = threadIdx.x >> 2;           // 0..63 (local l)
        const int d0 = (threadIdx.x & 3) * 16;     // 0,16,32,48
        const bf16* p = kvp + (size_t)(l0 + ll) * kstride + (size_t)b * 2 * EE + EE + h * DD + d0;
        v8bf r0 = *(const v8bf*)(p);
        v8bf r1 = *(const v8bf*)(p + 8);
        *(v8bf*)(&tile[ll * TSTR + d0])     = r0;
        *(v8bf*)(&tile[ll * TSTR + d0 + 8]) = r1;
    }
    __syncthreads();

    // write: 64 d rows x 64 L, coalesced along L
    {
        const int d  = threadIdx.x >> 2;           // 0..63
        const int ls = (threadIdx.x & 3) * 16;     // 0,16,32,48
        v8bf w0, w1;
#pragma unroll
        for (int j = 0; j < 8; ++j) {
            w0[j] = tile[(ls + j)     * TSTR + d];
            w1[j] = tile[(ls + 8 + j) * TSTR + d];
        }
        bf16* q = vT + ((size_t)bh * DD + d) * LK + l0 + ls;
        *(v8bf*)(q)     = w0;
        *(v8bf*)(q + 8) = w1;
    }
}

// ---------------------------------------------------------------------------
// Fused attention per (b, h, 16-row q tile):
//   scores = Qh @ Kh^T  (16 x 1024, K=64)  -> mask -> softmax
//   attn written to global (non-temporal streaming), P staged bf16 in LDS
//   ctx = P @ VhT  (16 x 64, K=1024) using the pre-transposed V panel
// 8 waves: phase 1 each wave owns 8 contiguous 16-wide k-column tiles;
// phase 2 waves are (d-tile 0..3) x (k-half 0..1), combined through LDS.
// ---------------------------------------------------------------------------
#define PSTR (1024 + 16)   // bf16 row stride for P: keeps 16B alignment, breaks bank conflicts

__global__ __launch_bounds__(256)
void attn_fused(const bf16* __restrict__ q_ws, const bf16* __restrict__ kv_ws,
                const bf16* __restrict__ vT,
                const unsigned char* __restrict__ mask,
                float* __restrict__ attn_out, bf16* __restrict__ ctx_ws)
{
    __shared__ bf16  Pbuf[16 * PSTR];
    __shared__ float redmax[8][16];
    __shared__ float redsum[8][16];
    __shared__ float ctxred[4][16][16];

    const int bid = blockIdx.x;
    const int qt  = bid & 63;          // 64 q tiles of 16 rows
    const int bh  = bid >> 6;          // b*H + h
    const int h   = bh & (HH - 1);
    const int b   = bh >> 4;
    const int q0  = qt << 4;

    const int lane = threadIdx.x & 31;
    const int wave = threadIdx.x >> 5;
    const int lr   = lane & 15;
    const int hi   = lane >> 4;

    const size_t qstride = (size_t)BB * EE;       // 8192
    const size_t kstride = (size_t)BB * 2 * EE;   // 16384
    const bf16* qbase = q_ws  + (size_t)b * EE + h * DD;
    const bf16* kbase = kv_ws + (size_t)b * 2 * EE + h * DD;
    const bf16* vtb   = vT + (size_t)bh * DD * LK;   // row d, stride LK

    // ---- Phase 1: scores = Qh @ Kh^T --------------------------------------
    V16 afrag[2];
#pragma unroll
    for (int c = 0; c < 2; ++c) {
        const bf16* p = qbase + (size_t)(q0 + lr) * qstride + c * 32 + hi * 8;
        afrag[c].h[0] = *(const v8bf*)(p);
        afrag[c].h[1] = *(const v8bf*)(p + 16);
    }

    v8f sc[8];
#pragma unroll
    for (int i = 0; i < 8; ++i) {
        const int kcol = (wave * 8 + i) * 16;
        v8f accv = {};
#pragma unroll
        for (int c = 0; c < 2; ++c) {
            V16 bfr;
            const bf16* p = kbase + (size_t)(kcol + lr) * kstride + c * 32 + hi * 8;
            bfr.h[0] = *(const v8bf*)(p);
            bfr.h[1] = *(const v8bf*)(p + 16);
            accv = __builtin_amdgcn_wmma_f32_16x16x32_bf16(
                false, afrag[c].v, false, bfr.v, (short)0, accv, false, false);
        }
        // mask: each lane owns one k-column of this tile
        if (mask[b * LK + kcol + lr]) {
#pragma unroll
            for (int r = 0; r < 8; ++r) accv[r] = -1e9f;
        }
        sc[i] = accv;
    }

    // ---- Softmax over k (rows hi*8+r live in this lane's VGPRs) -----------
    float pm[8];
#pragma unroll
    for (int r = 0; r < 8; ++r) {
        float m = sc[0][r];
#pragma unroll
        for (int i = 1; i < 8; ++i) m = fmaxf(m, sc[i][r]);
        pm[r] = m;
    }
#pragma unroll
    for (int off = 1; off < 16; off <<= 1)
#pragma unroll
        for (int r = 0; r < 8; ++r)
            pm[r] = fmaxf(pm[r], __shfl_xor(pm[r], off, 32));
    if (lr == 0) {
#pragma unroll
        for (int r = 0; r < 8; ++r) redmax[wave][hi * 8 + r] = pm[r];
    }
    __syncthreads();

    float rmax[8];
#pragma unroll
    for (int r = 0; r < 8; ++r) {
        float m = -3.0e38f;
#pragma unroll
        for (int w = 0; w < 8; ++w) m = fmaxf(m, redmax[w][hi * 8 + r]);
        rmax[r] = m;
    }

    float ps[8] = {0.f, 0.f, 0.f, 0.f, 0.f, 0.f, 0.f, 0.f};
#pragma unroll
    for (int i = 0; i < 8; ++i)
#pragma unroll
        for (int r = 0; r < 8; ++r) {
            const float e = __expf(sc[i][r] - rmax[r]);
            sc[i][r] = e;
            ps[r] += e;
        }
#pragma unroll
    for (int off = 1; off < 16; off <<= 1)
#pragma unroll
        for (int r = 0; r < 8; ++r)
            ps[r] += __shfl_xor(ps[r], off, 32);
    if (lr == 0) {
#pragma unroll
        for (int r = 0; r < 8; ++r) redsum[wave][hi * 8 + r] = ps[r];
    }
    __syncthreads();

    float rinv[8];
#pragma unroll
    for (int r = 0; r < 8; ++r) {
        float s = 0.f;
#pragma unroll
        for (int w = 0; w < 8; ++w) s += redsum[w][hi * 8 + r];
        rinv[r] = 1.0f / s;
    }

    // ---- Write attn (streaming) and stage P (bf16) in LDS -----------------
    float* arow = attn_out + ((size_t)bh * LQ + q0) * LK;
#pragma unroll
    for (int i = 0; i < 8; ++i) {
        const int col = (wave * 8 + i) * 16 + lr;
#pragma unroll
        for (int r = 0; r < 8; ++r) {
            const int qrow = hi * 8 + r;
            const float p = sc[i][r] * rinv[r];
            __builtin_nontemporal_store(p, &arow[(size_t)qrow * LK + col]);
            Pbuf[qrow * PSTR + col] = (bf16)p;
        }
    }
    __syncthreads();

    // ---- Phase 2: ctx = P @ VhT (16 x 64, K = 1024) -----------------------
    const int ntile = wave & 3;     // d tile
    const int khalf = wave >> 2;    // k half
    const int d0 = ntile * 16;

    v8f acc = {};
    for (int kc = khalf * 16; kc < khalf * 16 + 16; ++kc) {
        V16 a;
        const bf16* pp = &Pbuf[lr * PSTR + kc * 32 + hi * 8];
        a.h[0] = *(const v8bf*)(pp);
        a.h[1] = *(const v8bf*)(pp + 16);

        V16 bb;
        const bf16* pv = vtb + (size_t)(d0 + lr) * LK + kc * 32 + hi * 8;
        bb.h[0] = *(const v8bf*)(pv);
        bb.h[1] = *(const v8bf*)(pv + 16);

        acc = __builtin_amdgcn_wmma_f32_16x16x32_bf16(
            false, a.v, false, bb.v, (short)0, acc, false, false);
    }

    if (khalf == 1) {
#pragma unroll
        for (int r = 0; r < 8; ++r)
            ctxred[ntile][hi * 8 + r][lr] = acc[r];
    }
    __syncthreads();
    if (khalf == 0) {
#pragma unroll
        for (int r = 0; r < 8; ++r) {
            const int row = hi * 8 + r;
            const float v = acc[r] + ctxred[ntile][row][lr];
            ctx_ws[((size_t)(q0 + row) * BB + b) * EE + h * DD + d0 + lr] = (bf16)v;
        }
    }
}

// ---------------------------------------------------------------------------
// Host-side orchestration
// ---------------------------------------------------------------------------
extern "C" void kernel_launch(void* const* d_in, const int* in_sizes, int n_in,
                              void* d_out, int out_size, void* d_ws, size_t ws_size,
                              hipStream_t stream) {
    const float* query = (const float*)d_in[0];          // (LQ,B,E)
    const float* key   = (const float*)d_in[1];          // (LK,B,E)
    const unsigned char* attn_mask = (const unsigned char*)d_in[2]; // (B,LK) bool
    const float* wq  = (const float*)d_in[3];            // (E,E)
    const float* wkv = (const float*)d_in[4];            // (2E,E)
    const float* wo  = (const float*)d_in[5];            // (E,E)

    const int nQ  = LQ * BB * EE;     // 8388608
    const int nK  = LK * BB * EE;     // 8388608
    const int nWq = EE * EE;          // 1048576
    const int nWkv = 2 * EE * EE;     // 2097152
    const int nWo = EE * EE;

    // Workspace carve-up (bf16 buffers)
    char* ws = (char*)d_ws;
    size_t off = 0;
    auto carve = [&](size_t bytes) { char* p = ws + off; off += (bytes + 255) & ~(size_t)255; return p; };
    bf16* qbf   = (bf16*)carve((size_t)nQ * 2);
    bf16* kbf   = (bf16*)carve((size_t)nK * 2);
    bf16* wqbf  = (bf16*)carve((size_t)nWq * 2);
    bf16* wkvbf = (bf16*)carve((size_t)nWkv * 2);
    bf16* wobf  = (bf16*)carve((size_t)nWo * 2);
    bf16* qp    = (bf16*)carve((size_t)nQ * 2);               // scaled Q projection
    bf16* kvp   = (bf16*)carve((size_t)LK * BB * 2 * EE * 2); // KV projection
    bf16* vT    = (bf16*)carve((size_t)BB * HH * DD * LK * 2);// V transposed [b,h][d][LK]
    bf16* ctx   = (bf16*)carve((size_t)nQ * 2);               // attention context

    float* out_proj = (float*)d_out;                          // (LQ,B,E)
    float* attn_out = (float*)d_out + (size_t)LQ * BB * EE;   // (B,H,LQ,LK)

    // 1) convert inputs & weights to bf16
    cvt_f32_bf16<<<1024, 256, 0, stream>>>(query, qbf, nQ);
    cvt_f32_bf16<<<1024, 256, 0, stream>>>(key, kbf, nK);
    cvt_f32_bf16<<<256, 256, 0, stream>>>(wq, wqbf, nWq);
    cvt_f32_bf16<<<512, 256, 0, stream>>>(wkv, wkvbf, nWkv);
    cvt_f32_bf16<<<256, 256, 0, stream>>>(wo, wobf, nWo);

    // 2) projections (M = LQ*B = 8192 rows)
    const float scaling = 0.125f; // 1/sqrt(64)
    gemm_bf16_tn<<<dim3(8192 / 128, 1024 / 128), 256, 0, stream>>>(
        qbf, wqbf, qp, 8192, 1024, 1024, scaling, 0);
    gemm_bf16_tn<<<dim3(8192 / 128, 2048 / 128), 256, 0, stream>>>(
        kbf, wkvbf, kvp, 8192, 2048, 1024, 1.0f, 0);

    // 2b) transpose V for the P@V WMMA B-fragments
    transpose_v<<<dim3(BB * HH, LK / 64), 256, 0, stream>>>(kvp, vT);

    // 3) fused attention: B*H*(LQ/16) = 8192 blocks
    attn_fused<<<BB * HH * (LQ / 16), 256, 0, stream>>>(
        qp, kvp, vT, attn_mask, attn_out, ctx);

    // 4) output projection -> f32 directly into d_out
    gemm_bf16_tn<<<dim3(8192 / 128, 1024 / 128), 256, 0, stream>>>(
        ctx, wobf, out_proj, 8192, 1024, 1024, 1.0f, 1);

    (void)in_sizes; (void)n_in; (void)out_size; (void)ws_size;
}
